// PointNetSkeleton_91096256348438
// MI455X (gfx1250) — compile-verified, matrix operand order checked by reading, not run on hardware
//
#include <hip/hip_runtime.h>
#include <hip/hip_bf16.h>
#include <math.h>

typedef __attribute__((ext_vector_type(16))) _Float16 v16h;
typedef __attribute__((ext_vector_type(8)))  float    v8f;

#define B_CLOUDS 32
#define N1 1024
#define S1 512
#define S2 128
#define KNB 64

// ---------------------------------------------------------------------------
// Fragment helpers (layouts per CDNA5 ISA 7.12.2)
// ---------------------------------------------------------------------------
typedef union { v16h h; unsigned int u[8]; } AFragU;

// A: 16x32 f16 tile. lane<16 -> M=lane, K pairs {0..7} in v0..3, {16..23} in v4..7
//    lane>=16 -> M=lane-16, K pairs {8..15} and {24..31}. 2 halfs per VGPR.
__device__ __forceinline__ v16h load_a_frag(const _Float16* A, int rowBase,
                                            int strideHalfs, int kbase, int lane) {
    const int l15 = lane & 15;
    const int hi  = lane >> 4;
    const int row = rowBase + l15;
    const unsigned int* Au = (const unsigned int*)A;  // stride even -> 4B aligned
    const int rbase = row * strideHalfs + kbase + (hi ? 8 : 0);
    AFragU f;
#pragma unroll
    for (int v = 0; v < 8; ++v) {
        const int k = ((v < 4) ? 0 : 16) + 2 * (v & 3);
        f.u[v] = Au[(rbase + k) >> 1];
    }
    return f.h;
}

// ---------------------------------------------------------------------------
// Weight pre-pack: f32 [Cin][Cout] -> fragment-major f16, Cin padded to 32.
// Packed element p = (((kt*NT + nt)*32 + lane)*16 + (2v+j))
//   K = kt*32 + (v<4?0:16) + (lane>=16?8:0) + 2*(v&3) + j ;  N = nt*16 + (lane&15)
// ---------------------------------------------------------------------------
__global__ void pack_w_kernel(const float* __restrict__ W, _Float16* __restrict__ Wp,
                              int Cin, int CinPad, int Cout) {
    const int p = blockIdx.x * blockDim.x + threadIdx.x;
    const int total = CinPad * Cout;
    if (p >= total) return;
    const int NT   = Cout >> 4;
    const int h    = p & 15;
    const int lane = (p >> 4) & 31;
    const int rest = p >> 9;
    const int nt = rest % NT;
    const int kt = rest / NT;
    const int v = h >> 1, j = h & 1;
    const int K = kt * 32 + ((v < 4) ? 0 : 16) + ((lane >= 16) ? 8 : 0) + 2 * (v & 3) + j;
    const int Ncol = nt * 16 + (lane & 15);
    const float val = (K < Cin) ? W[(size_t)K * Cout + Ncol] : 0.f;
    Wp[p] = (_Float16)val;
}

// ---------------------------------------------------------------------------
// Farthest point sampling: one block (256 thr) per cloud; argmax with
// first-index tie-break (matches jnp.argmax), start at index 0.
// ---------------------------------------------------------------------------
__global__ __launch_bounds__(256)
void fps_kernel(const float* __restrict__ pos, int N, int S,
                int* __restrict__ idxOut, float* __restrict__ posq) {
    const int b = blockIdx.x, tid = threadIdx.x;
    __shared__ float d[1024];
    __shared__ float rv[256];
    __shared__ int   ri[256];
    __shared__ float cp[3];
    const float* P = pos + (size_t)b * N * 3;
    const float p0x = P[0], p0y = P[1], p0z = P[2];
    for (int i = tid; i < N; i += 256) {
        const float dx = P[i * 3 + 0] - p0x;
        const float dy = P[i * 3 + 1] - p0y;
        const float dz = P[i * 3 + 2] - p0z;
        d[i] = dx * dx + dy * dy + dz * dz;
    }
    if (tid == 0) {
        idxOut[(size_t)b * S] = 0;
        posq[(size_t)b * S * 3 + 0] = p0x;
        posq[(size_t)b * S * 3 + 1] = p0y;
        posq[(size_t)b * S * 3 + 2] = p0z;
    }
    __syncthreads();
    for (int s = 1; s < S; ++s) {
        float bv = -1.f; int bi = 0;
        for (int i = tid; i < N; i += 256) {
            const float v = d[i];
            if (v > bv) { bv = v; bi = i; }
        }
        rv[tid] = bv; ri[tid] = bi;
        __syncthreads();
        for (int off = 128; off > 0; off >>= 1) {
            if (tid < off) {
                const float ov = rv[tid + off]; const int oi = ri[tid + off];
                if (ov > rv[tid] || (ov == rv[tid] && oi < ri[tid])) { rv[tid] = ov; ri[tid] = oi; }
            }
            __syncthreads();
        }
        if (tid == 0) {
            const int i = ri[0];
            idxOut[(size_t)b * S + s] = i;
            cp[0] = P[i * 3 + 0]; cp[1] = P[i * 3 + 1]; cp[2] = P[i * 3 + 2];
            posq[((size_t)b * S + s) * 3 + 0] = cp[0];
            posq[((size_t)b * S + s) * 3 + 1] = cp[1];
            posq[((size_t)b * S + s) * 3 + 2] = cp[2];
        }
        __syncthreads();
        const float cx = cp[0], cy = cp[1], cz = cp[2];
        for (int i = tid; i < N; i += 256) {
            const float dx = P[i * 3 + 0] - cx;
            const float dy = P[i * 3 + 1] - cy;
            const float dz = P[i * 3 + 2] - cz;
            const float dd = dx * dx + dy * dy + dz * dz;
            d[i] = fminf(d[i], dd);
        }
        __syncthreads();
    }
}

// ---------------------------------------------------------------------------
// Ball query: first min(64,#valid) neighbors in ascending index order
// (matches reference top_k score trick); pad with 0 + nvalid count.
// ---------------------------------------------------------------------------
__global__ void neigh_kernel(const float* __restrict__ pos, const float* __restrict__ posq,
                             int N, int S, int total, float r2,
                             int* __restrict__ nidx, int* __restrict__ nvalid) {
    const int t = blockIdx.x * blockDim.x + threadIdx.x;
    if (t >= total) return;
    const int b = t / S;
    const float* P = pos + (size_t)b * N * 3;
    const float qx = posq[t * 3 + 0], qy = posq[t * 3 + 1], qz = posq[t * 3 + 2];
    int* nb = nidx + (size_t)t * KNB;
    int cnt = 0;
    for (int j = 0; j < N && cnt < KNB; ++j) {
        const float dx = P[j * 3 + 0] - qx;
        const float dy = P[j * 3 + 1] - qy;
        const float dz = P[j * 3 + 2] - qz;
        if (dx * dx + dy * dy + dz * dz < r2) nb[cnt++] = j;
    }
    nvalid[t] = cnt;
    for (int k = cnt; k < KNB; ++k) nb[k] = 0;
}

// ---------------------------------------------------------------------------
// LDS-resident WMMA GEMM layer: [64 x CIN] f16 (LDS) x packed W -> LDS f16.
// 4 waves, wave w owns rows 16w..16w+15.
// ---------------------------------------------------------------------------
template <int CIN, int COUT, bool RELU>
__device__ __forceinline__ void lds_gemm(const _Float16* inL, const _Float16* __restrict__ Wp,
                                         const float* __restrict__ bias, _Float16* outL, int tid) {
    constexpr int KT = CIN / 32, NT = COUT / 16;
    const int lane = tid & 31, wave = tid >> 5;
    const int mbase = wave * 16;
    const int l15 = lane & 15, hi = lane >> 4;
    const v16h* Bp = (const v16h*)Wp;
    for (int nt = 0; nt < NT; ++nt) {
        v8f acc = {0.f, 0.f, 0.f, 0.f, 0.f, 0.f, 0.f, 0.f};
#pragma unroll
        for (int kt = 0; kt < KT; ++kt) {
            const v16h a = load_a_frag(inL, mbase, CIN, kt * 32, lane);
            const v16h bf = Bp[(kt * NT + nt) * 32 + lane];
            acc = __builtin_amdgcn_wmma_f32_16x16x32_f16(false, a, false, bf,
                                                         (short)0, acc, false, false);
        }
        const int col = nt * 16 + l15;
        const float bv = bias[col];
#pragma unroll
        for (int v = 0; v < 8; ++v) {
            const int row = mbase + v + hi * 8;
            float val = acc[v] + bv;
            if (RELU) val = fmaxf(val, 0.f);
            outL[row * COUT + col] = (_Float16)val;
        }
    }
}

// ---------------------------------------------------------------------------
// Fused SA module: one block per (batch, center). Gather 64xCIN tile to LDS,
// 3 WMMA MLP layers in LDS, masked max over valid neighbors -> f32 out.
// ---------------------------------------------------------------------------
template <int CIN_PAD, int C1, int C2, int C3, int CX>
__global__ __launch_bounds__(128)
void sa_fused_kernel(const float* __restrict__ pos, const float* __restrict__ xin,
                     const float* __restrict__ posq, const int* __restrict__ nidx,
                     const int* __restrict__ nvalid,
                     const _Float16* __restrict__ Wp1, const _Float16* __restrict__ Wp2,
                     const _Float16* __restrict__ Wp3,
                     const float* __restrict__ b1, const float* __restrict__ b2,
                     const float* __restrict__ b3,
                     float* __restrict__ out, int N, int S) {
    constexpr int R0  = 64 * CIN_PAD * 2 + 64 * C1 * 2; // featA + buf1 (dead after L2)
    constexpr int R0b = 64 * C3 * 2;                    // buf3 (f16) aliases them
    constexpr int REGION0 = (R0 > R0b) ? R0 : R0b;
    __shared__ __align__(32) char lds[REGION0 + 64 * C2 * 2];
    _Float16* featA = (_Float16*)lds;
    _Float16* buf1  = (_Float16*)(lds + 64 * CIN_PAD * 2);
    _Float16* buf3  = (_Float16*)lds;
    _Float16* buf2  = (_Float16*)(lds + REGION0);

    const int tid = threadIdx.x;
    const int blk = blockIdx.x;
    const int b = blk / S;
    const int* nb = nidx + (size_t)blk * KNB;
    const int nv = nvalid[blk];
    const float q0 = posq[blk * 3 + 0];
    const float q1 = posq[blk * 3 + 1];
    const float q2 = posq[blk * 3 + 2];

    // gather features: [64 x CIN_PAD] = [x[nidx] | rel | zero-pad]
    for (int t = tid; t < 64 * CIN_PAD; t += 128) {
        const int k = t / CIN_PAD, c = t % CIN_PAD;
        const int j = nb[k];
        float v = 0.f;
        if (CX > 0 && c < CX) {
            v = xin[((size_t)b * N + j) * CX + c];
        } else if (c >= CX && c < CX + 3) {
            const int dcomp = c - CX;
            const float q = (dcomp == 0) ? q0 : ((dcomp == 1) ? q1 : q2);
            v = pos[((size_t)b * N + j) * 3 + dcomp] - q;
        }
        featA[t] = (_Float16)v;
    }
    __syncthreads();
    lds_gemm<CIN_PAD, C1, true >(featA, Wp1, b1, buf1, tid);
    __syncthreads();
    lds_gemm<C1,      C2, true >(buf1,  Wp2, b2, buf2, tid);
    __syncthreads();
    lds_gemm<C2,      C3, false>(buf2,  Wp3, b3, buf3, tid);
    __syncthreads();
    // masked max over valid neighbors (first nv rows)
    for (int c = tid; c < C3; c += 128) {
        float m = -3.0e38f;
        for (int k = 0; k < nv; ++k) m = fmaxf(m, (float)buf3[k * C3 + c]);
        out[(size_t)blk * C3 + c] = m;
    }
}

// ---------------------------------------------------------------------------
// Generic global-memory WMMA GEMM (SA3 layers): rows multiple of 64.
// grid = (rows/64, COUT/16), block = 128 (4 waves, one m-tile each).
// ---------------------------------------------------------------------------
template <int CIN, int COUT, bool RELU, bool OUT32>
__global__ __launch_bounds__(128)
void gemm_global_kernel(const _Float16* __restrict__ A, const _Float16* __restrict__ Wp,
                        const float* __restrict__ bias,
                        _Float16* __restrict__ out16, float* __restrict__ out32) {
    constexpr int KT = CIN / 32, NT = COUT / 16;
    const int lane = threadIdx.x & 31, wave = threadIdx.x >> 5;
    const int mbase = blockIdx.x * 64 + wave * 16;
    const int nt = blockIdx.y;
    const int l15 = lane & 15, hi = lane >> 4;
    const v16h* Bp = (const v16h*)Wp;
    v8f acc = {0.f, 0.f, 0.f, 0.f, 0.f, 0.f, 0.f, 0.f};
#pragma unroll
    for (int kt = 0; kt < KT; ++kt) {
        const v16h a = load_a_frag(A, mbase, CIN, kt * 32, lane);
        const v16h bf = Bp[(kt * NT + nt) * 32 + lane];
        acc = __builtin_amdgcn_wmma_f32_16x16x32_f16(false, a, false, bf,
                                                     (short)0, acc, false, false);
    }
    const int col = nt * 16 + l15;
    const float bv = bias[col];
#pragma unroll
    for (int v = 0; v < 8; ++v) {
        const int row = mbase + v + hi * 8;
        float val = acc[v] + bv;
        if (RELU) val = fmaxf(val, 0.f);
        if (OUT32) out32[(size_t)row * COUT + col] = val;
        else       out16[(size_t)row * COUT + col] = (_Float16)val;
    }
}

// SA3 input gather: feat3[4096][288] = [x2(256) | pos2(3) | pad]
__global__ void gather3_kernel(const float* __restrict__ x2, const float* __restrict__ pos2,
                               _Float16* __restrict__ feat) {
    const int t = blockIdx.x * blockDim.x + threadIdx.x;
    const int total = B_CLOUDS * S2 * 288;
    if (t >= total) return;
    const int row = t / 288, c = t % 288;
    float v = 0.f;
    if (c < 256)      v = x2[(size_t)row * 256 + c];
    else if (c < 259) v = pos2[(size_t)row * 3 + (c - 256)];
    feat[t] = (_Float16)v;
}

// global max over 128 rows per batch
__global__ void max3_kernel(const float* __restrict__ h, float* __restrict__ g) {
    const int t = blockIdx.x * blockDim.x + threadIdx.x;
    const int total = B_CLOUDS * 1024;
    if (t >= total) return;
    const int b = t / 1024, c = t % 1024;
    float m = -3.0e38f;
    for (int k = 0; k < S2; ++k) m = fmaxf(m, h[((size_t)b * S2 + k) * 1024 + c]);
    g[t] = m;
}

// head: 1024 -> 532 (relu) -> 40, log_softmax. One block per batch row.
__global__ __launch_bounds__(256)
void head_kernel(const float* __restrict__ g,
                 const float* __restrict__ W0, const float* __restrict__ b0,
                 const float* __restrict__ W1, const float* __restrict__ b1,
                 float* __restrict__ out) {
    const int b = blockIdx.x, tid = threadIdx.x;
    __shared__ float gl[1024];
    __shared__ float y1[532];
    __shared__ float y2[40];
    __shared__ float mm[2];
    for (int i = tid; i < 1024; i += 256) gl[i] = g[(size_t)b * 1024 + i];
    __syncthreads();
    for (int c = tid; c < 532; c += 256) {
        float acc = b0[c];
        for (int k = 0; k < 1024; ++k) acc += gl[k] * W0[(size_t)k * 532 + c];
        y1[c] = fmaxf(acc, 0.f);
    }
    __syncthreads();
    if (tid < 40) {
        float acc = b1[tid];
        for (int k = 0; k < 532; ++k) acc += y1[k] * W1[(size_t)k * 40 + tid];
        y2[tid] = acc;
    }
    __syncthreads();
    if (tid == 0) {
        float m = y2[0];
        for (int k = 1; k < 40; ++k) m = fmaxf(m, y2[k]);
        float s = 0.f;
        for (int k = 0; k < 40; ++k) s += expf(y2[k] - m);
        mm[0] = m; mm[1] = logf(s);
    }
    __syncthreads();
    if (tid < 40) out[b * 40 + tid] = y2[tid] - mm[0] - mm[1];
}

// ---------------------------------------------------------------------------
extern "C" void kernel_launch(void* const* d_in, const int* in_sizes, int n_in,
                              void* d_out, int out_size, void* d_ws, size_t ws_size,
                              hipStream_t stream) {
    (void)in_sizes; (void)n_in; (void)out_size; (void)ws_size;
    const float* pos = (const float*)d_in[0];
    const float* w[9]; const float* bs[9];
    for (int i = 0; i < 9; ++i) { w[i] = (const float*)d_in[1 + 2 * i]; bs[i] = (const float*)d_in[2 + 2 * i]; }
    const float* hw0 = (const float*)d_in[19];
    const float* hb0 = (const float*)d_in[20];
    const float* hw1 = (const float*)d_in[21];
    const float* hb1 = (const float*)d_in[22];
    float* out = (float*)d_out;

    char* base = (char*)d_ws;
    size_t off = 0;
    auto alloc = [&](size_t bytes) -> void* {
        off = (off + 255) & ~(size_t)255;
        void* p = base + off;
        off += bytes;
        return p;
    };

    int*       idx1    = (int*)alloc((size_t)B_CLOUDS * S1 * 4);
    float*     pos1    = (float*)alloc((size_t)B_CLOUDS * S1 * 3 * 4);
    int*       nidx1   = (int*)alloc((size_t)B_CLOUDS * S1 * KNB * 4);
    int*       nval1   = (int*)alloc((size_t)B_CLOUDS * S1 * 4);
    float*     x1      = (float*)alloc((size_t)B_CLOUDS * S1 * 128 * 4);
    int*       idx2    = (int*)alloc((size_t)B_CLOUDS * S2 * 4);
    float*     pos2    = (float*)alloc((size_t)B_CLOUDS * S2 * 3 * 4);
    int*       nidx2   = (int*)alloc((size_t)B_CLOUDS * S2 * KNB * 4);
    int*       nval2   = (int*)alloc((size_t)B_CLOUDS * S2 * 4);
    float*     x2      = (float*)alloc((size_t)B_CLOUDS * S2 * 256 * 4);
    _Float16*  feat3   = (_Float16*)alloc((size_t)B_CLOUDS * S2 * 288 * 2);
    _Float16*  h3a     = (_Float16*)alloc((size_t)B_CLOUDS * S2 * 256 * 2);
    _Float16*  h3b     = (_Float16*)alloc((size_t)B_CLOUDS * S2 * 512 * 2);
    float*     h3c     = (float*)alloc((size_t)B_CLOUDS * S2 * 1024 * 4);
    float*     g       = (float*)alloc((size_t)B_CLOUDS * 1024 * 4);
    // packed weights (Cin padded to mult of 32, f16)
    _Float16* Wp[9];
    const int cin[9]    = {3, 64, 64, 131, 128, 128, 259, 256, 512};
    const int cinpad[9] = {32, 64, 64, 160, 128, 128, 288, 256, 512};
    const int cout[9]   = {64, 64, 128, 128, 128, 256, 256, 512, 1024};
    for (int i = 0; i < 9; ++i) Wp[i] = (_Float16*)alloc((size_t)cinpad[i] * cout[i] * 2);

    // 1) pack all MLP weights to WMMA fragment-major layout
    for (int i = 0; i < 9; ++i) {
        const int elems = cinpad[i] * cout[i];
        pack_w_kernel<<<(elems + 255) / 256, 256, 0, stream>>>(w[i], Wp[i], cin[i], cinpad[i], cout[i]);
    }

    // 2) SA1: fps -> radius -> fused PointConv
    fps_kernel<<<B_CLOUDS, 256, 0, stream>>>(pos, N1, S1, idx1, pos1);
    {
        const int total = B_CLOUDS * S1;
        neigh_kernel<<<(total + 255) / 256, 256, 0, stream>>>(pos, pos1, N1, S1, total,
                                                              0.2f * 0.2f, nidx1, nval1);
        sa_fused_kernel<32, 64, 64, 128, 0><<<total, 128, 0, stream>>>(
            pos, nullptr, pos1, nidx1, nval1, Wp[0], Wp[1], Wp[2], bs[0], bs[1], bs[2],
            x1, N1, S1);
    }

    // 3) SA2
    fps_kernel<<<B_CLOUDS, 256, 0, stream>>>(pos1, S1, S2, idx2, pos2);
    {
        const int total = B_CLOUDS * S2;
        neigh_kernel<<<(total + 255) / 256, 256, 0, stream>>>(pos1, pos2, S1, S2, total,
                                                              0.4f * 0.4f, nidx2, nval2);
        sa_fused_kernel<160, 128, 128, 256, 128><<<total, 128, 0, stream>>>(
            pos1, x1, pos2, nidx2, nval2, Wp[3], Wp[4], Wp[5], bs[3], bs[4], bs[5],
            x2, S1, S2);
    }

    // 4) GlobalSA: MLP 259->256->512->1024 on 4096 rows, then per-batch max
    {
        const int rows = B_CLOUDS * S2; // 4096
        const int gtotal = rows * 288;
        gather3_kernel<<<(gtotal + 255) / 256, 256, 0, stream>>>(x2, pos2, feat3);
        gemm_global_kernel<288, 256, true,  false><<<dim3(rows / 64, 16), 128, 0, stream>>>(
            feat3, Wp[6], bs[6], h3a, nullptr);
        gemm_global_kernel<256, 512, true,  false><<<dim3(rows / 64, 32), 128, 0, stream>>>(
            h3a, Wp[7], bs[7], h3b, nullptr);
        gemm_global_kernel<512, 1024, false, true><<<dim3(rows / 64, 64), 128, 0, stream>>>(
            h3b, Wp[8], bs[8], nullptr, h3c);
        const int mtotal = B_CLOUDS * 1024;
        max3_kernel<<<(mtotal + 255) / 256, 256, 0, stream>>>(h3c, g);
    }

    // 5) head + log_softmax
    head_kernel<<<B_CLOUDS, 256, 0, stream>>>(g, hw0, hb0, hw1, hb1, out);
}
